// CogForge_7129645711723
// MI455X (gfx1250) — matrix operand order repaired and checked by reading, not compiled, not run on hardware
//
#include <hip/hip_runtime.h>

// ---------------------------------------------------------------------------
// CogForge GQA prefill (RoPE + sliding-window+global mask) for gfx1250.
// All GEMM-like math runs through v_wmma_f32_16x16x32_bf16 (wave32 WMMA).
// B-side operands (weights, V) are stored K-major so every WMMA operand
// assembles from contiguous dword loads (compiler fuses to global_load_b128).
// Q is pre-scaled by 1/sqrt(DH); interior key blocks skip masking entirely.
// ---------------------------------------------------------------------------

#define DEV __device__ __forceinline__

typedef unsigned short u16;
typedef __attribute__((ext_vector_type(16))) __bf16       v16bf;
typedef __attribute__((ext_vector_type(8)))  float        v8f;
typedef __attribute__((ext_vector_type(8)))  unsigned int v8u;

union BPack { v8u u; v16bf b; };

static constexpr int Bsz = 2, Tt = 2048, DM = 768, NH = 12, NKV = 4, DH = 64;
static constexpr int GROUPS = 3, WIN = 512, GLB = 64;
static constexpr int ROWS = Bsz * Tt;                 // 4096
static constexpr int NQKV = NH * DH + 2 * NKV * DH;   // 1280

DEV u16 f2bf(float f) {
    unsigned u = __float_as_uint(f);
    u += 0x7FFFu + ((u >> 16) & 1u);       // round-to-nearest-even
    return (u16)(u >> 16);
}
DEV float bf2f(u16 h) { return __uint_as_float(((unsigned)h) << 16); }

DEV v8f vzero() { v8f z = {0.f,0.f,0.f,0.f,0.f,0.f,0.f,0.f}; return z; }

// A-matrix (16x32 bf16) K index for VGPR-dword j, lane-half hi (contiguous: k = 16*hi + 2*j, matches B)
DEV int akk(int j, int hi) {
    int kb = hi << 4;
    return kb + 2 * j;
}

// ---------------------------------------------------------------------------
// fp32 -> bf16 streaming convert
__global__ __launch_bounds__(256)
void cvt_f2bf(const float* __restrict__ s, u16* __restrict__ d, int n) {
    int i = blockIdx.x * blockDim.x + threadIdx.x;
    if (i < n) d[i] = f2bf(s[i]);
}

// fp32 [R x C] -> bf16 transposed [C x R]
__global__ __launch_bounds__(256)
void cvt_f2bf_T(const float* __restrict__ s, u16* __restrict__ d, int R, int C) {
    int i = blockIdx.x * blockDim.x + threadIdx.x;
    if (i >= R * C) return;
    int c = i / R, r = i - c * R;          // i indexes the transposed output
    d[i] = f2bf(s[(size_t)r * C + c]);
}

// pack wq|wk|wv -> bf16, TRANSPOSED: [NQKV x DM] (K-major for WMMA B loads)
__global__ __launch_bounds__(256)
void pack_w_T(const float* __restrict__ wq, const float* __restrict__ wk,
              const float* __restrict__ wv, u16* __restrict__ d) {
    int i = blockIdx.x * blockDim.x + threadIdx.x;
    if (i >= DM * NQKV) return;
    int c = i / DM, r = i - c * DM;        // output element [c][r]  (c = N, r = K)
    float v;
    if (c < NH * DH)                 v = wq[(size_t)r * (NH * DH) + c];
    else if (c < NH * DH + NKV * DH) v = wk[(size_t)r * (NKV * DH) + (c - NH * DH)];
    else                             v = wv[(size_t)r * (NKV * DH) + (c - NH * DH - NKV * DH)];
    d[i] = f2bf(v);
}

// ---------------------------------------------------------------------------
// bf16 GEMM: D[M,N] = A[M,K] * B[K,N] with B stored transposed Bt[N,K].
// One wave computes a 16x64 tile; all operand loads are contiguous dwords.
template <bool OUT_BF16>
__global__ __launch_bounds__(256)
void gemm_bf16(const u16* __restrict__ A, const u16* __restrict__ Bt,
               void* __restrict__ outp, int M, int N, int K) {
    const int wave = (blockIdx.x * blockDim.x + threadIdx.x) >> 5;
    const int lane = threadIdx.x & 31;
    const int colGroups = N >> 6;
    const int row0 = (wave / colGroups) << 4;
    const int col0 = (wave % colGroups) << 6;
    if (row0 >= M) return;
    const int m   = lane & 15;
    const int hi  = lane >> 4;
    const int kbB = hi << 4;

    v8f acc[4];
#pragma unroll
    for (int c = 0; c < 4; ++c) acc[c] = vzero();

    const u16* arow = A + (size_t)(row0 + m) * K;
    const u16* brow[4];
#pragma unroll
    for (int c = 0; c < 4; ++c)
        brow[c] = Bt + (size_t)(col0 + (c << 4) + m) * K;

    for (int k0 = 0; k0 < K; k0 += 32) {
        if (k0 + 32 < K) {                     // gfx1250 global_prefetch_b8
            __builtin_prefetch(arow + k0 + 32, 0, 0);
            __builtin_prefetch(brow[0] + k0 + 32, 0, 0);
        }
        BPack a;
#pragma unroll
        for (int j = 0; j < 8; ++j)
            a.u[j] = *(const unsigned int*)(arow + k0 + akk(j, hi));
#pragma unroll
        for (int c = 0; c < 4; ++c) {
            BPack bm;
#pragma unroll
            for (int j = 0; j < 8; ++j)        // Bt row is contiguous along K
                bm.u[j] = *(const unsigned int*)(brow[c] + k0 + kbB + 2 * j);
            acc[c] = __builtin_amdgcn_wmma_f32_16x16x32_bf16(
                false, a.b, false, bm.b, (short)0, acc[c], false, false);
        }
    }
#pragma unroll
    for (int c = 0; c < 4; ++c)
#pragma unroll
        for (int r = 0; r < 8; ++r) {
            int row = row0 + r + (hi << 3);
            int col = col0 + (c << 4) + m;
            if (OUT_BF16) ((u16*)outp)[(size_t)row * N + col] = f2bf(acc[c][r]);
            else          ((float*)outp)[(size_t)row * N + col] = acc[c][r];
        }
}

// ---------------------------------------------------------------------------
// RoPE on q,k + split. q: [B,H,T,DH] PRE-SCALED by 1/sqrt(DH),
// k: [B,HKV,T,DH], v TRANSPOSED: [B,HKV,DH,T]
__global__ __launch_bounds__(256)
void rope_split(const u16* __restrict__ qkv, u16* __restrict__ qb,
                u16* __restrict__ kb, u16* __restrict__ vbT) {
    int idx = blockIdx.x * blockDim.x + threadIdx.x;
    if (idx >= ROWS * DH) return;
    int d   = idx & (DH - 1);
    int row = idx >> 6;           // b*T + t
    int t   = row & (Tt - 1);
    int b   = row >> 11;
    int j   = d & 31;
    float inv = __expf(-0.28782313662425572f * (float)j);  // 10000^(-2j/64)
    float ang = (float)t * inv;
    float cs = __cosf(ang), sn = __sinf(ang);
    const u16* rowp = qkv + (size_t)row * NQKV;
#pragma unroll
    for (int h = 0; h < NH; ++h) {
        const u16* qp = rowp + h * DH;
        float x0 = bf2f(qp[d]);
        float xr = (d < 32) ? -bf2f(qp[d + 32]) : bf2f(qp[d - 32]);
        // fold softmax scale 1/sqrt(64)=0.125 into Q (exact: power of two)
        qb[((size_t)(b * NH + h) * Tt + t) * DH + d] = f2bf((x0 * cs + xr * sn) * 0.125f);
    }
#pragma unroll
    for (int hk = 0; hk < NKV; ++hk) {
        const u16* kp = rowp + NH * DH + hk * DH;
        float x0 = bf2f(kp[d]);
        float xr = (d < 32) ? -bf2f(kp[d + 32]) : bf2f(kp[d - 32]);
        kb[((size_t)(b * NKV + hk) * Tt + t) * DH + d] = f2bf(x0 * cs + xr * sn);
        vbT[((size_t)(b * NKV + hk) * DH + d) * Tt + t] =
            rowp[NH * DH + NKV * DH + hk * DH + d];
    }
}

// ---------------------------------------------------------------------------
// Flash-style masked attention: one wave per (b, h, 16-query tile).
__global__ __launch_bounds__(256)
void attn_kernel(const u16* __restrict__ qb, const u16* __restrict__ kb,
                 const u16* __restrict__ vbT, u16* __restrict__ ob) {
    __shared__ u16 pshare[8][16 * 32];   // per-wave P staging (C->A re-layout)
    const int lane  = threadIdx.x & 31;
    const int wslot = threadIdx.x >> 5;
    const int wave  = (blockIdx.x << 3) + wslot;
    const int QT = Tt / 16;              // 128 query tiles
    const int qtile = wave % QT;
    const int h = (wave / QT) % NH;
    const int b = wave / (QT * NH);
    const int hk = h / GROUPS;
    const int q0 = qtile << 4;

    const u16* Q   = qb  + (size_t)(b * NH  + h ) * Tt * DH;
    const u16* Kp  = kb  + (size_t)(b * NKV + hk) * Tt * DH;
    const u16* VpT = vbT + (size_t)(b * NKV + hk) * DH * Tt;   // [DH][T]

    const int m   = lane & 15;
    const int hi  = lane >> 4;
    const int kbB = hi << 4;

    // Q tile in A layout, both K-halves of DH=64 (already scaled by 0.125)
    BPack qa[2];
    {
        const u16* qrow = Q + (size_t)(q0 + m) * DH;
#pragma unroll
        for (int t2 = 0; t2 < 2; ++t2)
#pragma unroll
            for (int j2 = 0; j2 < 8; ++j2)
                qa[t2].u[j2] = *(const unsigned int*)(qrow + t2 * 32 + akk(j2, hi));
    }

    v8f oacc[4];
#pragma unroll
    for (int c = 0; c < 4; ++c) oacc[c] = vzero();
    float rm[8], rl[8];
#pragma unroll
    for (int r = 0; r < 8; ++r) { rm[r] = -3.0e38f; rl[r] = 0.f; }

    u16* ps = pshare[wslot];

    auto process = [&](int jb) {
        // S = Qs @ K^T for 32 keys, as two 16-key C tiles
        v8f s[2]; s[0] = vzero(); s[1] = vzero();
#pragma unroll
        for (int sub = 0; sub < 2; ++sub)
#pragma unroll
            for (int kk2 = 0; kk2 < 2; ++kk2) {
                BPack bk;
                const u16* krow = Kp + (size_t)(jb + sub * 16 + m) * DH + kk2 * 32 + kbB;
#pragma unroll
                for (int j2 = 0; j2 < 8; ++j2)
                    bk.u[j2] = *(const unsigned int*)(krow + 2 * j2);
                s[sub] = __builtin_amdgcn_wmma_f32_16x16x32_bf16(
                    false, qa[kk2].b, false, bk.b, (short)0, s[sub], false, false);
            }
        // Interior blocks need no masking (wave-uniform test -> SALU branch):
        //  causal for all rows:      jb+31 <= q0
        //  inside window for all:    jb >= q0-(WIN-16)  OR fully global: jb+31 < GLB
        bool fullOK = (jb + 31 <= q0) && ((jb >= q0 - (WIN - 16)) || (jb + 31 < GLB));
        if (!fullOK) {
#pragma unroll
            for (int r = 0; r < 8; ++r) {
                int i = q0 + r + (hi << 3);
                int j0 = jb + m, j1 = jb + 16 + m;
                bool ok0 = (j0 <= i) && (((i - j0) < WIN) || (j0 < GLB));
                bool ok1 = (j1 <= i) && (((i - j1) < WIN) || (j1 < GLB));
                s[0][r] = ok0 ? s[0][r] : -3.0e38f;
                s[1][r] = ok1 ? s[1][r] : -3.0e38f;
            }
        }
        // block row-max (reduce across the 16-lane half)
        float bm[8];
#pragma unroll
        for (int r = 0; r < 8; ++r) {
            float v = fmaxf(s[0][r], s[1][r]);
#pragma unroll
            for (int mm = 8; mm >= 1; mm >>= 1) v = fmaxf(v, __shfl_xor(v, mm, 32));
            bm[r] = v;
        }
        // online softmax update; stage P (bf16) into LDS in C layout
#pragma unroll
        for (int r = 0; r < 8; ++r) {
            float nm   = fmaxf(rm[r], bm[r]);
            float corr = __expf(rm[r] - nm);
            float p0 = __expf(s[0][r] - nm);
            float p1 = __expf(s[1][r] - nm);
            float bs = p0 + p1;
#pragma unroll
            for (int mm = 8; mm >= 1; mm >>= 1) bs += __shfl_xor(bs, mm, 32);
            rl[r] = rl[r] * corr + bs;
            rm[r] = nm;
#pragma unroll
            for (int c = 0; c < 4; ++c) oacc[c][r] *= corr;
            int rowM = r + (hi << 3);
            ps[rowM * 32 + m]      = f2bf(p0);
            ps[rowM * 32 + 16 + m] = f2bf(p1);
        }
        // read P back in A layout (same-wave DS is in order; backend inserts waits)
        BPack pa;
#pragma unroll
        for (int j2 = 0; j2 < 8; ++j2)
            pa.u[j2] = *(const unsigned int*)(ps + m * 32 + akk(j2, hi));
        // O += P @ V : V^T rows are contiguous along keys -> dword loads
#pragma unroll
        for (int c = 0; c < 4; ++c) {
            BPack bv;
            const u16* vrow = VpT + (size_t)((c << 4) + m) * Tt + jb + kbB;
#pragma unroll
            for (int j2 = 0; j2 < 8; ++j2)
                bv.u[j2] = *(const unsigned int*)(vrow + 2 * j2);
            oacc[c] = __builtin_amdgcn_wmma_f32_16x16x32_bf16(
                false, pa.b, false, bv.b, (short)0, oacc[c], false, false);
        }
    };

    // global tokens [0, 64)
    for (int jb = 0; jb < GLB; jb += 32) process(jb);
    // sliding window (union over the 16 rows), never overlapping the global blocks
    int wlo = q0 - (WIN - 1);
    int start2 = (wlo <= GLB) ? GLB : (wlo & ~31);
    for (int jb = start2; jb < q0 + 16; jb += 32) process(jb);

    // normalize and store bf16 attention output [B*T, H*DH]
#pragma unroll
    for (int r = 0; r < 8; ++r) {
        float invl = 1.0f / rl[r];
        int t = q0 + r + (hi << 3);
#pragma unroll
        for (int c = 0; c < 4; ++c) {
            int col = h * DH + (c << 4) + m;
            ob[(size_t)(b * Tt + t) * (NH * DH) + col] = f2bf(oacc[c][r] * invl);
        }
    }
}

// ---------------------------------------------------------------------------
extern "C" void kernel_launch(void* const* d_in, const int* in_sizes, int n_in,
                              void* d_out, int out_size, void* d_ws, size_t ws_size,
                              hipStream_t stream) {
    (void)in_sizes; (void)n_in; (void)out_size; (void)ws_size;
    const float* x  = (const float*)d_in[0];
    const float* wq = (const float*)d_in[1];
    const float* wk = (const float*)d_in[2];
    const float* wv = (const float*)d_in[3];
    const float* wo = (const float*)d_in[4];
    float* out = (float*)d_out;

    char* ws = (char*)d_ws;                       // ~36.7 MB scratch, 256B-aligned regions
    u16* xb     = (u16*)(ws + 0);                 // 4096*768   bf16
    u16* wqkvT  = (u16*)(ws + 6291456);           // 1280*768   bf16 (transposed)
    u16* woT    = (u16*)(ws + 8257536);           // 768*768    bf16 (transposed)
    u16* qkvf   = (u16*)(ws + 9437184);           // 4096*1280  bf16
    u16* qbp    = (u16*)(ws + 19922944);          // [B,H,T,DH]   bf16 (scaled)
    u16* kbp    = (u16*)(ws + 26214400);          // [B,HKV,T,DH] bf16
    u16* vbpT   = (u16*)(ws + 28311552);          // [B,HKV,DH,T] bf16 (transposed)
    u16* obp    = (u16*)(ws + 30408704);          // 4096*768   bf16

    cvt_f2bf<<<(ROWS * DM + 255) / 256, 256, 0, stream>>>(x, xb, ROWS * DM);
    cvt_f2bf_T<<<(DM * DM + 255) / 256, 256, 0, stream>>>(wo, woT, DM, DM);
    pack_w_T<<<(DM * NQKV + 255) / 256, 256, 0, stream>>>(wq, wk, wv, wqkvT);

    // QKV projection: [4096x768] @ [768x1280]  -> 5120 waves -> 640 blocks
    gemm_bf16<true><<<640, 256, 0, stream>>>(xb, wqkvT, qkvf, ROWS, NQKV, DM);

    rope_split<<<(ROWS * DH + 255) / 256, 256, 0, stream>>>(qkvf, qbp, kbp, vbpT);

    // attention: B*H*(T/16) = 3072 waves -> 384 blocks of 8 waves
    attn_kernel<<<(Bsz * NH * (Tt / 16)) / 8, 256, 0, stream>>>(qbp, kbp, vbpT, obp);

    // output projection: [4096x768] @ [768x768] -> 3072 waves -> 384 blocks
    gemm_bf16<false><<<384, 256, 0, stream>>>(obp, woT, out, ROWS, DM, DM);
}
